// GRUFromScratch_8443905704172
// MI455X (gfx1250) — matrix operand order repaired and checked by reading, not compiled, not run on hardware
//
#include <hip/hip_runtime.h>

typedef __attribute__((ext_vector_type(16))) __bf16 v16bf;
typedef __attribute__((ext_vector_type(8)))  float  v8f;
typedef unsigned short u16;
typedef unsigned int   u32;

namespace gru {
constexpr int S = 512, B = 64, I = 256, H = 1024, O = 128;
constexpr int MT  = B / 16;          // 4 M-tiles of 16 rows
constexpr int KTX = I / 32;          // 8  k-tiles from x
constexpr int KTH = H / 32;          // 32 k-tiles from hidden
constexpr int KT0 = (I + H) / 32;    // 40 k-tiles layer0
constexpr int KT1 = (2 * H) / 32;    // 64 k-tiles layer1
constexpr int NTH = H / 16;          // 64 n-tiles
constexpr int NTO = O / 16;          // 8  n-tiles output
constexpr int TILE = 512;            // u16 per 16x32 tile (32 lanes * 16 elems)
constexpr int NWG = 64;              // persistent workgroups, 8 waves each

constexpr size_t SZ_HPK = (size_t)MT * KTH * TILE * 2;   // 128 KB packed hidden
constexpr size_t SZ_HF  = (size_t)B * H * 4;             // 256 KB fp32 hidden
// zero-initialized region: [barrier(1024B) | h0pk | h1pk | h0f32 | h1f32]
constexpr size_t OFF_BAR  = 0;
constexpr size_t OFF_H0PK = 1024;
constexpr size_t OFF_H1PK = OFF_H0PK + SZ_HPK;
constexpr size_t OFF_H0F  = OFF_H1PK + SZ_HPK;
constexpr size_t OFF_H1F  = OFF_H0F + SZ_HF;
constexpr size_t ZERO_BYTES = OFF_H1F + SZ_HF;           // 787456 (multiple of 1024)
constexpr size_t OFF_RCPK = ZERO_BYTES;                  // packed r*h staging
constexpr size_t OFF_Z    = OFF_RCPK + SZ_HPK;           // fp32 z gate
constexpr size_t OFF_XPK  = OFF_Z + SZ_HF;               // packed bf16 x (all steps)
constexpr size_t SZ_XPK   = (size_t)S * MT * KTX * TILE * 2;
constexpr size_t OFF_W0   = OFF_XPK + SZ_XPK;            // packed layer0 weights r,z,h
constexpr size_t SZ_W0    = (size_t)3 * NTH * KT0 * TILE * 2;
constexpr size_t OFF_W1   = OFF_W0 + SZ_W0;              // packed layer1 weights r,z,h
constexpr size_t SZ_W1    = (size_t)3 * NTH * KT1 * TILE * 2;
constexpr size_t OFF_WO   = OFF_W1 + SZ_W1;              // packed W_out
constexpr size_t SZ_WO    = (size_t)NTO * KTH * TILE * 2;
constexpr size_t WS_NEED  = OFF_WO + SZ_WO;              // ~38.7 MB
} // namespace gru

// fp32 -> bf16 round-to-nearest-even
__device__ __forceinline__ u16 f2bf(float f) {
  u32 u = __float_as_uint(f);
  u32 r = u + 0x7FFFu + ((u >> 16) & 1u);
  return (u16)(r >> 16);
}

// A-fragment (16x32, 16-bit) lane/element index for matrix element (M, Kk):
// lanes 0-15 hold M=0..15 with K pairs {0,1},{2,3},{4,5},{6,7},{16,17}...{22,23}
// lanes 16-31 hold the same rows with K offset +8. (ISA 7.12.2)
__device__ __forceinline__ int apack_idx(int M, int Kk) {
  int Lhi  = (Kk & 8) ? 1 : 0;
  int kp   = Kk - 8 * Lhi;
  int j    = kp & 1;
  int base = kp - j;                               // 0,2,4,6,16,18,20,22
  int v    = (base < 8) ? (base >> 1) : (((base - 16) >> 1) + 4);
  return (M + 16 * Lhi) * 16 + (2 * v + j);
}

// B-fragment (32x16, 16-bit) lane/element index for element (N, Kk):
// same K order as A: lanes 0-15 K=0..7,16..23; lanes 16-31 K=8..15,24..31; column N = lane%16.
__device__ __forceinline__ int bpack_idx(int N, int Kk) {
  return (N + 16 * ((Kk >> 3) & 1)) * 16 + (Kk & 7) + ((Kk >> 4) << 3);
}

// ---------------- pack kernel: fp32 -> packed bf16 fragments, zero state ----
__global__ void gru_pack_kernel(const float* __restrict__ x,
                                const float* __restrict__ Wr0, const float* __restrict__ Wz0,
                                const float* __restrict__ Wh0,
                                const float* __restrict__ Wr1, const float* __restrict__ Wz1,
                                const float* __restrict__ Wh1,
                                const float* __restrict__ Wout,
                                unsigned char* __restrict__ ws) {
  using namespace gru;
  u16* xpk = (u16*)(ws + OFF_XPK);
  u16* w0  = (u16*)(ws + OFF_W0);
  u16* w1  = (u16*)(ws + OFF_W1);
  u16* wo  = (u16*)(ws + OFF_WO);
  u32* zr  = (u32*)ws;

  const long K0 = I + H, K1 = 2 * H;
  const long N0 = 3L * H * K0;          // layer0 weights
  const long N1 = 3L * H * K1;          // layer1 weights
  const long N2 = (long)O * H;          // W_out
  const long N3 = (long)S * B * I;      // x
  const long NZ = (long)(ZERO_BYTES / 4);
  const long total = N0 + N1 + N2 + N3 + NZ;

  const long stride = (long)gridDim.x * blockDim.x;
  for (long e = (long)blockIdx.x * blockDim.x + threadIdx.x; e < total; e += stride) {
    long r = e;
    if (r < N0) {
      int g = (int)(r / (H * K0));
      long q = r % (H * K0);
      int n = (int)(q / K0), k = (int)(q % K0);
      const float* W = (g == 0) ? Wr0 : ((g == 1) ? Wz0 : Wh0);
      w0[((size_t)(g * NTH + (n >> 4)) * KT0 + (k >> 5)) * TILE + bpack_idx(n & 15, k & 31)] =
          f2bf(W[(size_t)n * K0 + k]);
      continue;
    }
    r -= N0;
    if (r < N1) {
      int g = (int)(r / (H * K1));
      long q = r % (H * K1);
      int n = (int)(q / K1), k = (int)(q % K1);
      const float* W = (g == 0) ? Wr1 : ((g == 1) ? Wz1 : Wh1);
      w1[((size_t)(g * NTH + (n >> 4)) * KT1 + (k >> 5)) * TILE + bpack_idx(n & 15, k & 31)] =
          f2bf(W[(size_t)n * K1 + k]);
      continue;
    }
    r -= N1;
    if (r < N2) {
      int n = (int)(r / H), k = (int)(r % H);
      wo[((size_t)(n >> 4) * KTH + (k >> 5)) * TILE + bpack_idx(n & 15, k & 31)] =
          f2bf(Wout[(size_t)n * H + k]);
      continue;
    }
    r -= N2;
    if (r < N3) {
      int t = (int)(r / (B * I));
      int q = (int)(r % (B * I));
      int m = q / I, i = q % I;
      xpk[((size_t)(t * MT + (m >> 4)) * KTX + (i >> 5)) * TILE + apack_idx(m & 15, i & 31)] =
          f2bf(x[r]);
      continue;
    }
    r -= N3;
    zr[r] = 0u;  // barrier words, h0/h1 packed + fp32 state
  }
}

// ---------------- persistent GRU kernel -----------------------------------
__device__ __forceinline__ void grid_barrier(u32* bar) {
  __syncthreads();
  if (threadIdx.x == 0) {
    __threadfence();
    u32 g = __hip_atomic_load(&bar[1], __ATOMIC_RELAXED, __HIP_MEMORY_SCOPE_AGENT);
    u32 a = __hip_atomic_fetch_add(&bar[0], 1u, __ATOMIC_ACQ_REL, __HIP_MEMORY_SCOPE_AGENT);
    if (a == (u32)gru::NWG - 1u) {
      __hip_atomic_store(&bar[0], 0u, __ATOMIC_RELAXED, __HIP_MEMORY_SCOPE_AGENT);
      __hip_atomic_store(&bar[1], g + 1u, __ATOMIC_RELEASE, __HIP_MEMORY_SCOPE_AGENT);
    } else {
      while (__hip_atomic_load(&bar[1], __ATOMIC_ACQUIRE, __HIP_MEMORY_SCOPE_AGENT) == g) {
        __builtin_amdgcn_s_sleep(1);
      }
    }
    __threadfence();
  }
  __syncthreads();
}

// K-loop over two concatenated A sources (packed fragments) against streamed B.
__device__ __forceinline__ v8f mm_tiles(const u16* __restrict__ A1, int n1,
                                        const u16* __restrict__ A2, int n2,
                                        const u16* __restrict__ Bw, int lane, v8f acc) {
  using namespace gru;
  const u16* a1 = A1 + lane * 16;
  const u16* a2 = A2 + lane * 16;
  const u16* b  = Bw + lane * 16;
  for (int kt = 0; kt < n1; ++kt) {
    __builtin_prefetch(b + (size_t)(kt + 1) * TILE, 0, 1);   // global_prefetch_b8
    v16bf av = *(const v16bf*)(a1 + (size_t)kt * TILE);
    v16bf bv = *(const v16bf*)(b + (size_t)kt * TILE);
    acc = __builtin_amdgcn_wmma_f32_16x16x32_bf16(false, av, false, bv, (short)0, acc,
                                                  false, false);
  }
  b += (size_t)n1 * TILE;
  for (int kt = 0; kt < n2; ++kt) {
    __builtin_prefetch(b + (size_t)(kt + 1) * TILE, 0, 1);
    v16bf av = *(const v16bf*)(a2 + (size_t)kt * TILE);
    v16bf bv = *(const v16bf*)(b + (size_t)kt * TILE);
    acc = __builtin_amdgcn_wmma_f32_16x16x32_bf16(false, av, false, bv, (short)0, acc,
                                                  false, false);
  }
  return acc;
}

__global__ __launch_bounds__(256) void gru_persistent_kernel(
    unsigned char* __restrict__ ws,
    const float* __restrict__ br0, const float* __restrict__ bz0, const float* __restrict__ bh0,
    const float* __restrict__ br1, const float* __restrict__ bz1, const float* __restrict__ bh1,
    const float* __restrict__ bo, float* __restrict__ out) {
  using namespace gru;
  u32* bar       = (u32*)(ws + OFF_BAR);
  const u16* xpk = (const u16*)(ws + OFF_XPK);
  const u16* w0  = (const u16*)(ws + OFF_W0);
  const u16* w1  = (const u16*)(ws + OFF_W1);
  const u16* wo  = (const u16*)(ws + OFF_WO);
  u16* h0pk      = (u16*)(ws + OFF_H0PK);
  u16* h1pk      = (u16*)(ws + OFF_H1PK);
  u16* rcpk      = (u16*)(ws + OFF_RCPK);
  float* h0f     = (float*)(ws + OFF_H0F);
  float* h1f     = (float*)(ws + OFF_H1F);
  float* zb      = (float*)(ws + OFF_Z);

  const int lane = threadIdx.x & 31;
  const int wave = threadIdx.x >> 5;
  const int job  = blockIdx.x * 8 + wave;  // 0..511
  const int hi   = lane >> 4;              // C-frag: M = v + 8*hi
  const int Ncol = lane & 15;              // C-frag: N = lane%16

  for (int t = 0; t < S; ++t) {
    const u16* xt = xpk + (size_t)t * MT * KTX * TILE;

    // ---- phase RZ0: r0 (fused r*h0 pack) and z0; 512 jobs ----
    {
      const int gate = job >> 8;           // 0=r, 1=z
      const int rem = job & 255;
      const int nt = rem >> 2, mt = rem & 3;
      v8f acc = {};
      acc = mm_tiles(xt + (size_t)mt * KTX * TILE, KTX,
                     h0pk + (size_t)mt * KTH * TILE, KTH,
                     w0 + (size_t)(gate * NTH + nt) * KT0 * TILE, lane, acc);
      const int n = nt * 16 + Ncol;
      const float bias = (gate == 0 ? br0 : bz0)[n];
      if (gate == 0) {
        const int ktk = n >> 5, Kk = n & 31;
        for (int v = 0; v < 8; ++v) {
          const int m = mt * 16 + v + 8 * hi;
          const float s = 1.f / (1.f + __expf(-(acc[v] + bias)));
          rcpk[(size_t)(mt * KTH + ktk) * TILE + apack_idx(v + 8 * hi, Kk)] =
              f2bf(s * h0f[(size_t)m * H + n]);
        }
      } else {
        for (int v = 0; v < 8; ++v) {
          const int m = mt * 16 + v + 8 * hi;
          zb[(size_t)m * H + n] = 1.f / (1.f + __expf(-(acc[v] + bias)));
        }
      }
    }
    grid_barrier(bar);

    // ---- phase Htilde0 + h0 update; 256 jobs ----
    if (job < 256) {
      const int nt = job >> 2, mt = job & 3;
      v8f acc = {};
      acc = mm_tiles(xt + (size_t)mt * KTX * TILE, KTX,
                     rcpk + (size_t)mt * KTH * TILE, KTH,
                     w0 + (size_t)(2 * NTH + nt) * KT0 * TILE, lane, acc);
      const int n = nt * 16 + Ncol;
      const float bias = bh0[n];
      const int ktk = n >> 5, Kk = n & 31;
      for (int v = 0; v < 8; ++v) {
        const int m = mt * 16 + v + 8 * hi;
        const float ht = tanhf(acc[v] + bias);
        const float z = zb[(size_t)m * H + n];
        const float hn = (1.f - z) * h0f[(size_t)m * H + n] + z * ht;
        h0f[(size_t)m * H + n] = hn;
        h0pk[(size_t)(mt * KTH + ktk) * TILE + apack_idx(v + 8 * hi, Kk)] = f2bf(hn);
      }
    }
    grid_barrier(bar);

    // ---- phase RZ1: r1 (fused r*h1 pack) and z1; 512 jobs ----
    {
      const int gate = job >> 8;
      const int rem = job & 255;
      const int nt = rem >> 2, mt = rem & 3;
      v8f acc = {};
      acc = mm_tiles(h0pk + (size_t)mt * KTH * TILE, KTH,
                     h1pk + (size_t)mt * KTH * TILE, KTH,
                     w1 + (size_t)(gate * NTH + nt) * KT1 * TILE, lane, acc);
      const int n = nt * 16 + Ncol;
      const float bias = (gate == 0 ? br1 : bz1)[n];
      if (gate == 0) {
        const int ktk = n >> 5, Kk = n & 31;
        for (int v = 0; v < 8; ++v) {
          const int m = mt * 16 + v + 8 * hi;
          const float s = 1.f / (1.f + __expf(-(acc[v] + bias)));
          rcpk[(size_t)(mt * KTH + ktk) * TILE + apack_idx(v + 8 * hi, Kk)] =
              f2bf(s * h1f[(size_t)m * H + n]);
        }
      } else {
        for (int v = 0; v < 8; ++v) {
          const int m = mt * 16 + v + 8 * hi;
          zb[(size_t)m * H + n] = 1.f / (1.f + __expf(-(acc[v] + bias)));
        }
      }
    }
    grid_barrier(bar);

    // ---- phase Htilde1 + h1 update; 256 jobs ----
    if (job < 256) {
      const int nt = job >> 2, mt = job & 3;
      v8f acc = {};
      acc = mm_tiles(h0pk + (size_t)mt * KTH * TILE, KTH,
                     rcpk + (size_t)mt * KTH * TILE, KTH,
                     w1 + (size_t)(2 * NTH + nt) * KT1 * TILE, lane, acc);
      const int n = nt * 16 + Ncol;
      const float bias = bh1[n];
      const int ktk = n >> 5, Kk = n & 31;
      for (int v = 0; v < 8; ++v) {
        const int m = mt * 16 + v + 8 * hi;
        const float ht = tanhf(acc[v] + bias);
        const float z = zb[(size_t)m * H + n];
        const float hn = (1.f - z) * h1f[(size_t)m * H + n] + z * ht;
        h1f[(size_t)m * H + n] = hn;
        h1pk[(size_t)(mt * KTH + ktk) * TILE + apack_idx(v + 8 * hi, Kk)] = f2bf(hn);
      }
    }
    grid_barrier(bar);

    // ---- phase OUT: out_t = h1 @ Wout^T + b_out; 32 jobs ----
    if (job < 32) {
      const int nt = job >> 2, mt = job & 3;
      v8f acc = {};
      acc = mm_tiles(h1pk + (size_t)mt * KTH * TILE, KTH,
                     h1pk, 0,
                     wo + (size_t)nt * KTH * TILE, lane, acc);
      const int n = nt * 16 + Ncol;
      const float bias = bo[n];
      for (int v = 0; v < 8; ++v) {
        const int m = mt * 16 + v + 8 * hi;
        out[((size_t)t * B + m) * O + n] = acc[v] + bias;
      }
    }
    grid_barrier(bar);
  }

  // final hidden state stack [2, B, H] appended after [S, B, O]
  float* tail = out + (size_t)S * B * O;
  for (int e = blockIdx.x * blockDim.x + threadIdx.x; e < 2 * B * H; e += NWG * 256) {
    tail[e] = (e < B * H) ? h0f[e] : h1f[e - B * H];
  }
}

extern "C" void kernel_launch(void* const* d_in, const int* in_sizes, int n_in,
                              void* d_out, int out_size, void* d_ws, size_t ws_size,
                              hipStream_t stream) {
  (void)in_sizes; (void)n_in; (void)out_size;
  if (ws_size < gru::WS_NEED) return;  // scratch too small: do nothing (deterministic)

  const float* x    = (const float*)d_in[0];
  const float* Wr0  = (const float*)d_in[1];
  const float* br0  = (const float*)d_in[2];
  const float* Wz0  = (const float*)d_in[3];
  const float* bz0  = (const float*)d_in[4];
  const float* Wh0  = (const float*)d_in[5];
  const float* bh0  = (const float*)d_in[6];
  const float* Wr1  = (const float*)d_in[7];
  const float* br1  = (const float*)d_in[8];
  const float* Wz1  = (const float*)d_in[9];
  const float* bz1  = (const float*)d_in[10];
  const float* Wh1  = (const float*)d_in[11];
  const float* bh1  = (const float*)d_in[12];
  const float* Wout = (const float*)d_in[13];
  const float* bo   = (const float*)d_in[14];
  float* out = (float*)d_out;
  unsigned char* ws = (unsigned char*)d_ws;

  gru_pack_kernel<<<4096, 256, 0, stream>>>(x, Wr0, Wz0, Wh0, Wr1, Wz1, Wh1, Wout, ws);
  gru_persistent_kernel<<<gru::NWG, 256, 0, stream>>>(ws, br0, bz0, bh0, br1, bz1, bh1,
                                                      bo, out);
}
